// SelfAttention_5463198401158
// MI455X (gfx1250) — compile-verified, matrix operand order checked by reading, not run on hardware
//
#include <hip/hip_runtime.h>
#include <hip/hip_bf16.h>

// MI455X / gfx1250 GQA attention layer.
// B=4 S=512 D=4096, HQ=32 HKV=8 HD=128, start_pos=1024, T=1536.
// Matmuls: v_wmma_f32_16x16x32_bf16 (wave32).  Tile staging: Tensor Data
// Mover (tensor_load_to_lds, TENSORcnt) with double buffering in the GEMM.

typedef unsigned short u16;
typedef __attribute__((ext_vector_type(16))) __bf16 v16bf;
typedef __attribute__((ext_vector_type(8)))  float  v8f;
typedef __attribute__((ext_vector_type(4)))  unsigned v4u;
typedef __attribute__((ext_vector_type(8)))  int    v8i_t;
typedef __attribute__((ext_vector_type(4)))  int    v4i_t;

union frag16 { v16bf v; uint4 q[2]; };
union pack8  { uint4 q; u16 s[8]; };

__device__ __forceinline__ u16 f2bf(float f) {
    union { float f; unsigned u; } x; x.f = f;
    unsigned r = x.u + 0x7FFFu + ((x.u >> 16) & 1u);   // round-to-nearest-even
    return (u16)(r >> 16);
}

#if __has_builtin(__builtin_amdgcn_tensor_load_to_lds)
#define HAVE_TDM 1
#else
#define HAVE_TDM 0
#endif

#if HAVE_TDM
// 2D TDM load: tile_x elements (2B each) x tile_y rows -> LDS at lds_off with
// pad_enable so each row lands on a padded stride.  D# per CDNA5 ISA §8.
// This toolchain exposes the 6-arg builtin:
//   (uint32x4 g0, int32x8 g1, int32x4 g2, int32x4 g3, int32x8, int cpol)
__device__ __forceinline__ void tdm_load_2d(
    const void* gaddr, unsigned lds_off,
    unsigned tile_x, unsigned tile_y,
    unsigned dim0, unsigned dim1, unsigned stride0,
    unsigned pad_interval, unsigned pad_amount)
{
    unsigned long long ga = (unsigned long long)gaddr;
    v4u g0;
    g0[0] = 1u;                                        // count=1, user desc
    g0[1] = lds_off;                                   // lds_addr
    g0[2] = (unsigned)ga;                              // global_addr[31:0]
    g0[3] = (unsigned)((ga >> 32) & 0x1FFFFFFull) | (2u << 30);  // type=2
    v8i_t g1;
    g1[0] = (int)((1u << 16)                 // data_size = 2 bytes
                | (1u << 20)                 // pad_enable
                | (pad_interval << 22)
                | (pad_amount << 25));
    g1[1] = (int)((dim0 & 0xFFFFu) << 16);                       // tensor_dim0 lo
    g1[2] = (int)(((dim0 >> 16) & 0xFFFFu) | ((dim1 & 0xFFFFu) << 16));
    g1[3] = (int)(((dim1 >> 16) & 0xFFFFu) | ((tile_x & 0xFFFFu) << 16));
    g1[4] = (int)(tile_y & 0xFFFFu);                             // tile_dim1
    g1[5] = (int)stride0;                                        // dim0_stride lo
    g1[6] = 0;
    g1[7] = 0;
    v4i_t z4 = {0, 0, 0, 0};
    v8i_t z8 = {0, 0, 0, 0, 0, 0, 0, 0};
    __builtin_amdgcn_tensor_load_to_lds(g0, g1, z4, z4, z8, 0);
}
__device__ __forceinline__ unsigned lds_off_of(const void* p) {
    return (unsigned)(unsigned long long)p;            // addrspace(3) offset
}
#endif

// ---------------------------------------------------------------------------
// fp32 -> bf16 streaming conversion (8 elements / thread).
// ---------------------------------------------------------------------------
__global__ __launch_bounds__(256) void cvt_bf16_kernel(
    const float* __restrict__ in, u16* __restrict__ out, int n8)
{
    int i = blockIdx.x * 256 + threadIdx.x;
    if (i >= n8) return;
    float4 f0 = ((const float4*)in)[2 * i];
    float4 f1 = ((const float4*)in)[2 * i + 1];
    pack8 p;
    p.s[0]=f2bf(f0.x); p.s[1]=f2bf(f0.y); p.s[2]=f2bf(f0.z); p.s[3]=f2bf(f0.w);
    p.s[4]=f2bf(f1.x); p.s[5]=f2bf(f1.y); p.s[6]=f2bf(f1.z); p.s[7]=f2bf(f1.w);
    ((uint4*)out)[i] = p.q;
}

// ---------------------------------------------------------------------------
// GEMM: C[M,N] (f32) = A[M,K] x B[N,K]^T, A/B bf16.  Block tile 128x128,
// 8 waves (4Mx2N), wave tile 32x64, K-step 32.  TDM double-buffered staging.
// ---------------------------------------------------------------------------
__global__ __launch_bounds__(256) void gemm_bf16wmma(
    const u16* __restrict__ A, const u16* __restrict__ Bw,
    float* __restrict__ C, int M, int N, int K)
{
    __shared__ __align__(16) u16 As[2][128 * 40];
    __shared__ __align__(16) u16 Bs[2][128 * 40];

    const int tid  = threadIdx.x;
    const int lane = tid & 31;
    const int wave = tid >> 5;
    const int wM   = wave >> 1;          // 0..3
    const int wN   = wave & 1;           // 0..1
    const int half = lane >> 4;          // 0/1
    const int ln   = lane & 15;
    const int m0   = blockIdx.y * 128;
    const int n0   = blockIdx.x * 128;

    const u16* Abase = A  + (size_t)m0 * K;
    const u16* Bbase = Bw + (size_t)n0 * K;

    v8f acc[2][4] = {};

    // Prologue: stage k0=0 into buffer 0.
#if HAVE_TDM
    if (wave == 0) {
        // 32x64B rows, pad 16B per 64B -> 40-short LDS stride.
        tdm_load_2d(Abase, lds_off_of(&As[0][0]), 32, 128,
                    (unsigned)K, (unsigned)M, (unsigned)K, 3, 3);
        tdm_load_2d(Bbase, lds_off_of(&Bs[0][0]), 32, 128,
                    (unsigned)K, (unsigned)N, (unsigned)K, 3, 3);
        __builtin_amdgcn_s_wait_tensorcnt(0);
    }
#else
    for (int i = tid; i < 512; i += 256) {
        int row = i >> 2, seg = i & 3;
        *(uint4*)(As[0] + row * 40 + seg * 8) =
            *(const uint4*)(Abase + (size_t)row * K + seg * 8);
        *(uint4*)(Bs[0] + row * 40 + seg * 8) =
            *(const uint4*)(Bbase + (size_t)row * K + seg * 8);
    }
#endif
    __syncthreads();

    int cur = 0;
    for (int k0 = 0; k0 < K; k0 += 32) {
        const int nxt = cur ^ 1;
        if (k0 + 32 < K) {
#if HAVE_TDM
            if (wave == 0) {
                tdm_load_2d(Abase + k0 + 32, lds_off_of(&As[nxt][0]), 32, 128,
                            (unsigned)K, (unsigned)M, (unsigned)K, 3, 3);
                tdm_load_2d(Bbase + k0 + 32, lds_off_of(&Bs[nxt][0]), 32, 128,
                            (unsigned)K, (unsigned)N, (unsigned)K, 3, 3);
            }
#else
            for (int i = tid; i < 512; i += 256) {
                int row = i >> 2, seg = i & 3;
                *(uint4*)(As[nxt] + row * 40 + seg * 8) =
                    *(const uint4*)(Abase + (size_t)row * K + k0 + 32 + seg * 8);
                *(uint4*)(Bs[nxt] + row * 40 + seg * 8) =
                    *(const uint4*)(Bbase + (size_t)row * K + k0 + 32 + seg * 8);
            }
#endif
        }

        // A-frag (16x32 bf16): lane half selects K {0..7,16..23} vs {8..15,24..31}.
        frag16 a[2], b[4];
        for (int i = 0; i < 2; ++i) {
            int row = wM * 32 + i * 16 + ln;
            a[i].q[0] = *(const uint4*)(As[cur] + row * 40 + 8 * half);
            a[i].q[1] = *(const uint4*)(As[cur] + row * 40 + 8 * half + 16);
        }
        for (int j = 0; j < 4; ++j) {
            int row = wN * 64 + j * 16 + ln;
            b[j].q[0] = *(const uint4*)(Bs[cur] + row * 40 + 8 * half);
            b[j].q[1] = *(const uint4*)(Bs[cur] + row * 40 + 8 * half + 16);
        }
        for (int i = 0; i < 2; ++i)
            for (int j = 0; j < 4; ++j)
                acc[i][j] = __builtin_amdgcn_wmma_f32_16x16x32_bf16(
                    false, a[i].v, false, b[j].v, (short)0, acc[i][j], false, false);

#if HAVE_TDM
        if (wave == 0) __builtin_amdgcn_s_wait_tensorcnt(0);
#endif
        __syncthreads();
        cur = nxt;
    }

    // D layout: vgpr v + 8*half = row, lane&15 = col.
    for (int i = 0; i < 2; ++i)
        for (int j = 0; j < 4; ++j)
            for (int v = 0; v < 8; ++v) {
                int row = m0 + wM * 32 + i * 16 + v + 8 * half;
                int col = n0 + wN * 64 + j * 16 + ln;
                C[(size_t)row * N + col] = acc[i][j][v];
            }
}

// ---------------------------------------------------------------------------
// RoPE on Q: qf32 [b,s,h(32),d(128)] -> q_bf16 [b,h,s,d]. One pair per thread.
// ---------------------------------------------------------------------------
__global__ __launch_bounds__(256) void rope_q_kernel(
    const float* __restrict__ q, const float* __restrict__ fc,
    const float* __restrict__ fs, u16* __restrict__ out)
{
    int idx = blockIdx.x * 256 + threadIdx.x;        // 2^22
    int d2 = idx & 63;
    int s  = (idx >> 6) & 511;
    int h  = (idx >> 15) & 31;
    int b  = idx >> 20;
    size_t base = ((size_t)(b * 512 + s) * 4096) + h * 128;
    float xr = q[base + 2 * d2], xi = q[base + 2 * d2 + 1];
    float c = fc[s * 64 + d2], sn = fs[s * 64 + d2];
    size_t ob = ((size_t)(b * 32 + h) * 512 + s) * 128;
    out[ob + 2 * d2]     = f2bf(xr * c - xi * sn);
    out[ob + 2 * d2 + 1] = f2bf(xr * sn + xi * c);
}

// ---------------------------------------------------------------------------
// K cache build: t<1024 from cache_k fp32 [b,2048,8,128]; t>=1024 =
// RoPE(kf32 [b,s,8,128]).  Out: bf16 [b,kh,1536,128].
// ---------------------------------------------------------------------------
__global__ __launch_bounds__(256) void build_k_kernel(
    const float* __restrict__ cache_k, const float* __restrict__ knew,
    const float* __restrict__ fc, const float* __restrict__ fs,
    u16* __restrict__ out)
{
    int idx = blockIdx.x * 256 + threadIdx.x;        // 4*8*1536*64
    int d2 = idx & 63;
    int r  = idx >> 6;
    int t  = r % 1536; r /= 1536;
    int kh = r & 7;
    int b  = r >> 3;
    float xr, xi;
    if (t < 1024) {
        size_t src = ((size_t)((b * 2048 + t) * 8 + kh)) * 128;
        xr = cache_k[src + 2 * d2];
        xi = cache_k[src + 2 * d2 + 1];
    } else {
        int s = t - 1024;
        size_t src = (size_t)(b * 512 + s) * 1024 + kh * 128;
        float ar = knew[src + 2 * d2], ai = knew[src + 2 * d2 + 1];
        float c = fc[s * 64 + d2], sn = fs[s * 64 + d2];
        xr = ar * c - ai * sn;
        xi = ar * sn + ai * c;
    }
    size_t ob = ((size_t)(b * 8 + kh) * 1536 + t) * 128;
    out[ob + 2 * d2]     = f2bf(xr);
    out[ob + 2 * d2 + 1] = f2bf(xi);
}

__global__ __launch_bounds__(256) void build_v_kernel(
    const float* __restrict__ cache_v, const float* __restrict__ vnew,
    u16* __restrict__ out)
{
    int idx = blockIdx.x * 256 + threadIdx.x;        // 4*8*1536*128
    int d = idx & 127;
    int r = idx >> 7;
    int t = r % 1536; r /= 1536;
    int kh = r & 7;
    int b  = r >> 3;
    float x;
    if (t < 1024) x = cache_v[((size_t)((b * 2048 + t) * 8 + kh)) * 128 + d];
    else          x = vnew[(size_t)(b * 512 + (t - 1024)) * 1024 + kh * 128 + d];
    out[((size_t)(b * 8 + kh) * 1536 + t) * 128 + d] = f2bf(x);
}

// ---------------------------------------------------------------------------
// Flash attention, WMMA bf16.  Block = 128 thr (4 waves) = one (b,h) x 64 q
// rows; each wave owns 16 q rows.  K tile staged by TDM; V staged transposed;
// online softmax with half-wave shfl reductions.  Output bf16 [b,s,h,d].
// ---------------------------------------------------------------------------
__global__ __launch_bounds__(128) void attn_kernel(
    const u16* __restrict__ qbf, const u16* __restrict__ kall,
    const u16* __restrict__ vall, const float* __restrict__ masks,
    u16* __restrict__ out)
{
    __shared__ __align__(16) u16 kt[32 * 136];       // K tile, row-major padded
    __shared__ __align__(16) u16 vt[128 * 40];       // V tile, transposed [d][key]
    __shared__ __align__(16) u16 pt[4][16 * 40];     // per-wave P tile

    const int tid  = threadIdx.x;
    const int lane = tid & 31;
    const int wave = tid >> 5;
    const int half = lane >> 4, ln = lane & 15;
    const int qb = blockIdx.x;           // 0..7
    const int h  = blockIdx.y;           // 0..31
    const int b  = blockIdx.z;           // 0..3
    const int kh = h >> 2;
    const int q0 = qb * 64 + wave * 16;  // wave's q-row base (0..511)
    const float SCALE = 0.08838834764831845f;   // 1/sqrt(128)

    // Q fragments: 16 rows x 128 d = 4 chunks of K=32.
    const u16* qrow = qbf + (((size_t)(b * 32 + h) * 512) + q0 + ln) * 128;
    frag16 qf[4];
    for (int c = 0; c < 4; ++c) {
        qf[c].q[0] = *(const uint4*)(qrow + 32 * c + 8 * half);
        qf[c].q[1] = *(const uint4*)(qrow + 32 * c + 8 * half + 16);
    }

    const u16* kbase = kall + (size_t)(b * 8 + kh) * 1536 * 128;
    const u16* vbase = vall + (size_t)(b * 8 + kh) * 1536 * 128;

    v8f acc[8] = {};
    float m_[8], l_[8];
    for (int v = 0; v < 8; ++v) { m_[v] = -1e30f; l_[v] = 0.f; }

    for (int t = 0; t < 1536; t += 32) {
        __syncthreads();
        // Stage K tile 32x128 (row stride 136: 256B data + 16B pad).
#if HAVE_TDM
        if (wave == 0)
            tdm_load_2d(kbase + (size_t)t * 128, lds_off_of(&kt[0]), 128, 32,
                        128u, 1536u, 128u, 5, 3);
#else
        for (int i = tid; i < 512; i += 128) {
            int row = i >> 4, seg = i & 15;
            *(uint4*)(kt + row * 136 + seg * 8) =
                *(const uint4*)(kbase + (size_t)(t + row) * 128 + seg * 8);
        }
#endif
        // Stage V tile transposed: vt[d*40 + key].
        for (int i = tid; i < 512; i += 128) {
            int row = i >> 4, seg = i & 15;
            pack8 p; p.q = *(const uint4*)(vbase + (size_t)(t + row) * 128 + seg * 8);
            for (int e = 0; e < 8; ++e) vt[(seg * 8 + e) * 40 + row] = p.s[e];
        }
#if HAVE_TDM
        if (wave == 0) __builtin_amdgcn_s_wait_tensorcnt(0);
#endif
        __syncthreads();

        // Scores: two 16(q)x16(key) tiles, contraction over d in 4 chunks.
        v8f s[2] = {};
        for (int j = 0; j < 2; ++j)
            for (int c = 0; c < 4; ++c) {
                frag16 kf;
                int row = 16 * j + ln;
                kf.q[0] = *(const uint4*)(kt + row * 136 + 32 * c + 8 * half);
                kf.q[1] = *(const uint4*)(kt + row * 136 + 32 * c + 8 * half + 16);
                s[j] = __builtin_amdgcn_wmma_f32_16x16x32_bf16(
                    false, qf[c].v, false, kf.v, (short)0, s[j], false, false);
            }
        for (int j = 0; j < 2; ++j)
            for (int v = 0; v < 8; ++v)
                s[j][v] = s[j][v] * SCALE +
                          masks[(size_t)(q0 + v + 8 * half) * 1536 + t + 16 * j + ln];

        // Online softmax: row = v + 8*half; 16 cols live in 16 lanes of the
        // same half-wave -> shfl_xor reductions over offsets 1,2,4,8.
        float alpha[8];
        for (int v = 0; v < 8; ++v) {
            float mx = fmaxf(s[0][v], s[1][v]);
            for (int off = 1; off < 16; off <<= 1)
                mx = fmaxf(mx, __shfl_xor(mx, off, 32));
            float mnew = fmaxf(m_[v], mx);
            alpha[v] = __expf(m_[v] - mnew);
            float p0 = __expf(s[0][v] - mnew);
            float p1 = __expf(s[1][v] - mnew);
            s[0][v] = p0; s[1][v] = p1;
            float rs = p0 + p1;
            for (int off = 1; off < 16; off <<= 1)
                rs += __shfl_xor(rs, off, 32);
            l_[v] = l_[v] * alpha[v] + rs;
            m_[v] = mnew;
        }

        // P (16x32) -> per-wave LDS (score layout -> A-frag layout transpose).
        for (int j = 0; j < 2; ++j)
            for (int v = 0; v < 8; ++v)
                pt[wave][(v + 8 * half) * 40 + 16 * j + ln] = f2bf(s[j][v]);
        __syncthreads();

        frag16 af;
        af.q[0] = *(const uint4*)(pt[wave] + ln * 40 + 8 * half);
        af.q[1] = *(const uint4*)(pt[wave] + ln * 40 + 8 * half + 16);

        // Rescale accumulators then accumulate P x V over 8 d-tiles.
        for (int dt = 0; dt < 8; ++dt)
            for (int v = 0; v < 8; ++v)
                acc[dt][v] *= alpha[v];
        for (int dt = 0; dt < 8; ++dt) {
            frag16 vf;
            int d = 16 * dt + ln;
            vf.q[0] = *(const uint4*)(vt + d * 40 + 8 * half);
            vf.q[1] = *(const uint4*)(vt + d * 40 + 8 * half + 16);
            acc[dt] = __builtin_amdgcn_wmma_f32_16x16x32_bf16(
                false, af.v, false, vf.v, (short)0, acc[dt], false, false);
        }
    }

    // Normalize and write bf16 attention output [b,s,h,d] = [2048,4096].
    for (int v = 0; v < 8; ++v) {
        float inv = 1.f / l_[v];
        int row = b * 512 + q0 + v + 8 * half;
        for (int dt = 0; dt < 8; ++dt)
            out[(size_t)row * 4096 + h * 128 + dt * 16 + ln] = f2bf(acc[dt][v] * inv);
    }
}

// ---------------------------------------------------------------------------
// Host launcher. Workspace layout (bytes), ~160 MB total:
//   [0,     16.8M)  xbf   bf16 x            -> reused as attn_bf16 output
//   [16.8M, 50.3M)  w1bf  bf16 wq, later wo
//   [50.3M, 58.7M)  wkbf
//   [58.7M, 67.1M)  wvbf
//   [67.1M,100.7M)  qf32
//   [100.7M,109.1M) kf32
//   [109.1M,117.4M) vf32
//   [117.4M,134.2M) qbf  [b,h,s,d]
//   [134.2M,146.8M) kall bf16 [b,kh,t,d]
//   [146.8M,159.4M) vall bf16
// ---------------------------------------------------------------------------
extern "C" void kernel_launch(void* const* d_in, const int* in_sizes, int n_in,
                              void* d_out, int out_size, void* d_ws, size_t ws_size,
                              hipStream_t stream) {
    const float* x       = (const float*)d_in[0];
    const float* wq      = (const float*)d_in[1];
    const float* wk      = (const float*)d_in[2];
    const float* wv      = (const float*)d_in[3];
    const float* wo      = (const float*)d_in[4];
    const float* fcos    = (const float*)d_in[5];
    const float* fsin    = (const float*)d_in[6];
    const float* cache_k = (const float*)d_in[7];
    const float* cache_v = (const float*)d_in[8];
    const float* masks   = (const float*)d_in[9];
    float* out = (float*)d_out;

    char* ws = (char*)d_ws;
    u16*   xbf  = (u16*)  (ws);
    u16*   w1bf = (u16*)  (ws + 16777216);
    u16*   wkbf = (u16*)  (ws + 50331648);
    u16*   wvbf = (u16*)  (ws + 58720256);
    float* qf32 = (float*)(ws + 67108864);
    float* kf32 = (float*)(ws + 100663296);
    float* vf32 = (float*)(ws + 109051904);
    u16*   qbf  = (u16*)  (ws + 117440512);
    u16*   kall = (u16*)  (ws + 134217728);
    u16*   vall = (u16*)  (ws + 146800640);
    u16*   abf  = xbf;    // reuse: x is dead after the projection GEMMs

    // One-time bf16 conversion of activations + weights.
    cvt_bf16_kernel<<<4096, 256, 0, stream>>>(x,  xbf,  1048576);
    cvt_bf16_kernel<<<8192, 256, 0, stream>>>(wq, w1bf, 2097152);
    cvt_bf16_kernel<<<2048, 256, 0, stream>>>(wk, wkbf, 524288);
    cvt_bf16_kernel<<<2048, 256, 0, stream>>>(wv, wvbf, 524288);

    // Projections (M=2048, K=4096).
    gemm_bf16wmma<<<dim3(32, 16), 256, 0, stream>>>(xbf, w1bf, qf32, 2048, 4096, 4096);
    gemm_bf16wmma<<<dim3(8, 16),  256, 0, stream>>>(xbf, wkbf, kf32, 2048, 1024, 4096);
    gemm_bf16wmma<<<dim3(8, 16),  256, 0, stream>>>(xbf, wvbf, vf32, 2048, 1024, 4096);

    // wq's slot is free now; stage wo into it for the final projection.
    cvt_bf16_kernel<<<8192, 256, 0, stream>>>(wo, w1bf, 2097152);

    // RoPE + cache materialization in bf16.
    rope_q_kernel <<<16384, 256, 0, stream>>>(qf32, fcos, fsin, qbf);
    build_k_kernel<<<12288, 256, 0, stream>>>(cache_k, kf32, fcos, fsin, kall);
    build_v_kernel<<<24576, 256, 0, stream>>>(cache_v, vf32, vall);

    // Flash attention: grid (qblock=8, head=32, batch=4), 128 thr.
    attn_kernel<<<dim3(8, 32, 4), 128, 0, stream>>>(qbf, kall, vall, masks, abf);

    // Output projection.
    gemm_bf16wmma<<<dim3(32, 16), 256, 0, stream>>>(abf, w1bf, out, 2048, 4096, 4096);
}
